// GATLayer_1151051235523
// MI455X (gfx1250) — compile-verified, matrix operand order checked by reading, not run on hardware
//
#include <hip/hip_runtime.h>
#include <hip/hip_bf16.h>

// Problem constants (match the JAX reference)
#define BB    4
#define NN    2048
#define CIN   256
#define HH    4
#define CH    64
#define COUT  256
#define ALPHA 0.2f
#define JT    (NN / 32)   // 64 j-tiles of 32

typedef __attribute__((ext_vector_type(16))) __bf16          v16bf;
typedef __attribute__((ext_vector_type(8)))  float           v8f;

__device__ __forceinline__ float lrelu(float x) { return fmaxf(x, ALPHA * x); }

// K-slot base for 16-bit A/B operands of v_wmma_*_16x16x32_* (wave32):
// vgpr v, lane-half -> first of a contiguous K pair
__device__ __forceinline__ int k0_of(int v, int half) {
    return ((v >> 2) << 4) + (half << 3) + ((v & 3) << 1);
}

// Pack 8 consecutive-K floats (two float4s) into elements [base..base+7] of a
// bf16 WMMA operand vector. Hardware f32->bf16 convert (RNE).
__device__ __forceinline__ void pack8(v16bf& dst, int base, float4 q0, float4 q1) {
    dst[base + 0] = (__bf16)q0.x; dst[base + 1] = (__bf16)q0.y;
    dst[base + 2] = (__bf16)q0.z; dst[base + 3] = (__bf16)q0.w;
    dst[base + 4] = (__bf16)q1.x; dst[base + 5] = (__bf16)q1.y;
    dst[base + 6] = (__bf16)q1.z; dst[base + 7] = (__bf16)q1.w;
}

// ---------------------------------------------------------------------------
// Kernel 1: feats = X (8192x256) * W^T (256x256) + b, fp32 out, bf16 WMMA
// One wave per 16x16 tile; operands loaded as b128 runs (k pairs contiguous).
// ---------------------------------------------------------------------------
__global__ void feats_gemm(const float* __restrict__ x,
                           const float* __restrict__ W,
                           const float* __restrict__ bias,
                           float* __restrict__ feats) {
    int wave = (blockIdx.x * blockDim.x + threadIdx.x) >> 5;
    int lane = threadIdx.x & 31;
    int mt   = wave >> 4;      // 0..511  (row tile over B*N)
    int nt   = wave & 15;      // 0..15   (col tile over COUT)
    int m    = lane & 15;
    int half = lane >> 4;
    int arow = mt * 16 + m;          // A row for this lane
    int bcol = nt * 16 + m;          // B col (output channel o) for this lane

    v8f acc = {};
    for (int kk = 0; kk < CIN; kk += 32) {
        const float* xp = x + (size_t)arow * CIN + kk + half * 8;
        const float* wp = W + (size_t)bcol * CIN + kk + half * 8;
        float4 xa0 = *(const float4*)(xp);
        float4 xa1 = *(const float4*)(xp + 4);
        float4 xa2 = *(const float4*)(xp + 16);
        float4 xa3 = *(const float4*)(xp + 20);
        float4 wa0 = *(const float4*)(wp);
        float4 wa1 = *(const float4*)(wp + 4);
        float4 wa2 = *(const float4*)(wp + 16);
        float4 wa3 = *(const float4*)(wp + 20);
        v16bf av, bv;
        pack8(av, 0, xa0, xa1); pack8(av, 8, xa2, xa3);
        pack8(bv, 0, wa0, wa1); pack8(bv, 8, wa2, wa3);
        acc = __builtin_amdgcn_wmma_f32_16x16x32_bf16(false, av, false, bv,
                                                      (short)0, acc, false, false);
    }
    float bval = bias[bcol];
#pragma unroll
    for (int r = 0; r < 8; ++r) {
        int orow = mt * 16 + r + 8 * half;   // C layout: vgpr r -> M=r (lo), r+8 (hi)
        feats[(size_t)orow * COUT + bcol] = acc[r] + bval;
    }
}

// ---------------------------------------------------------------------------
// Kernel 2: per-node/head logits ls, ld (stored transposed: [b][h][n])
// ---------------------------------------------------------------------------
__global__ void logits_kernel(const float* __restrict__ feats,
                              const float* __restrict__ a,
                              float* __restrict__ lsT,
                              float* __restrict__ ldT) {
    int id = blockIdx.x * blockDim.x + threadIdx.x;   // B*N*H = 32768
    if (id >= BB * NN * HH) return;
    int h = id & (HH - 1);
    int n = (id >> 2) & (NN - 1);
    int b = id >> 13;
    const float* f = feats + (size_t)(b * NN + n) * COUT + h * CH;
    float s = 0.f, d = 0.f;
#pragma unroll 8
    for (int c = 0; c < CH; ++c) {
        float fv = f[c];
        s += fv * a[h * 2 * CH + c];
        d += fv * a[h * 2 * CH + CH + c];
    }
    lsT[(b * HH + h) * NN + n] = s;
    ldT[(b * HH + h) * NN + n] = d;
}

// ---------------------------------------------------------------------------
// Kernel 3: gmax[b][h] = max_n ld[b][h][n]  (safe softmax bound; lrelu monotone)
// ---------------------------------------------------------------------------
__global__ void gmax_kernel(const float* __restrict__ ldT, float* __restrict__ gmax) {
    __shared__ float red[256];
    int bh = blockIdx.x;                 // 16 blocks
    float m = -INFINITY;
    for (int n = threadIdx.x; n < NN; n += 256) m = fmaxf(m, ldT[bh * NN + n]);
    red[threadIdx.x] = m;
    __syncthreads();
    for (int s = 128; s > 0; s >>= 1) {
        if ((int)threadIdx.x < s) red[threadIdx.x] = fmaxf(red[threadIdx.x], red[threadIdx.x + s]);
        __syncthreads();
    }
    if (threadIdx.x == 0) gmax[bh] = red[0];
}

// ---------------------------------------------------------------------------
// Kernel 4: swizzle feats into bf16 B-operand lane layout:
// flat id = ((((b*JT+jt)*H+h)*4+cc)*32+lane)*8+v  -> pair (j=jt*32+k0, j+1) at col c
// so a lane's 16 bf16 B elements are 32 contiguous bytes.
// ---------------------------------------------------------------------------
__global__ void swz_kernel(const float* __restrict__ feats, unsigned short* __restrict__ fsw) {
    unsigned id = blockIdx.x * blockDim.x + threadIdx.x;  // 1,048,576
    int v    = id & 7;
    int lane = (id >> 3) & 31;
    int cc   = (id >> 8) & 3;
    int h    = (id >> 10) & 3;
    int jt   = (id >> 12) & 63;
    int b    = id >> 18;
    int half = lane >> 4;
    int n    = lane & 15;
    int k0   = k0_of(v, half);
    int j    = jt * 32 + k0;
    int c    = h * CH + cc * 16 + n;
    float f0 = feats[(size_t)(b * NN + j) * COUT + c];
    float f1 = feats[(size_t)(b * NN + j + 1) * COUT + c];
    __bf16* bp = (__bf16*)fsw;
    bp[(size_t)id * 2]     = (__bf16)f0;
    bp[(size_t)id * 2 + 1] = (__bf16)f1;
}

// ---------------------------------------------------------------------------
// Kernel 5: pack adjacency to bitmask: 32x traffic reduction (67MB -> 2MB)
// ---------------------------------------------------------------------------
__global__ void adjbits_kernel(const int* __restrict__ adj, unsigned* __restrict__ bits) {
    int id = blockIdx.x * blockDim.x + threadIdx.x;  // B*N*(N/32) = 524288
    int w = id & 63;
    int i = (id >> 6) & (NN - 1);
    int b = id >> 17;
    const int* row = adj + (size_t)(b * NN + i) * NN + w * 32;
    unsigned word = 0;
#pragma unroll 8
    for (int t = 0; t < 32; ++t) word |= (unsigned)(row[t] & 1) << t;
    bits[id] = word;
}

// ---------------------------------------------------------------------------
// Kernel 6: fused masked-softmax attention + P*feats via bf16 WMMA.
// Grid: B*(N/16) blocks, 128 threads (4 waves = 4 heads).
// Each wave: 16 rows x all 2048 j x one head x 64 out channels.
// ---------------------------------------------------------------------------
__global__ void attn_kernel(const unsigned* __restrict__ adjbits,
                            const float* __restrict__ lsT,
                            const float* __restrict__ ldT,
                            const float* __restrict__ gmax,
                            const unsigned short* __restrict__ fsw,
                            float* __restrict__ out) {
    __shared__ float s_lds[HH * 16];
    int h    = threadIdx.x >> 5;         // wave id == head
    int lane = threadIdx.x & 31;
    int b    = blockIdx.x >> 7;          // gridDim.x = B * (N/16) = 512
    int it   = blockIdx.x & 127;
    int i0   = it * 16;
    int m    = lane & 15;
    int half = lane >> 4;

    const float LOG2E = 1.4426950408889634f;
    float ls_m = lsT[(b * HH + h) * NN + i0 + m];
    float Mrow = lrelu(ls_m + gmax[b * HH + h]);     // >= any unmasked logit in row

    const float*    ldp   = ldT + (size_t)(b * HH + h) * NN;
    const unsigned* abase = adjbits + (size_t)(b * NN + i0 + m) * JT;
    const v16bf*    fbase = (const v16bf*)fsw;

    v8f acc0 = {}, acc1 = {}, acc2 = {}, acc3 = {};
    float rsum = 0.f;

    for (int jt = 0; jt < JT; ++jt) {
        unsigned aw = abase[jt];

        // lane's 16 ld_j values: two runs of 8 consecutive floats -> 4x b128
        const float* lp = ldp + jt * 32 + half * 8;
        float4 d0 = *(const float4*)(lp);
        float4 d1 = *(const float4*)(lp + 4);
        float4 d2 = *(const float4*)(lp + 16);
        float4 d3 = *(const float4*)(lp + 20);
        float dv[16] = { d0.x, d0.y, d0.z, d0.w, d1.x, d1.y, d1.z, d1.w,
                         d2.x, d2.y, d2.z, d2.w, d3.x, d3.y, d3.z, d3.w };
        v16bf A;
#pragma unroll
        for (int e = 0; e < 16; ++e) {
            int k = half * 8 + e + ((e < 8) ? 0 : 8);   // bit index within 32-j tile
            float l = lrelu(ls_m + dv[e]);
#if __has_builtin(__builtin_amdgcn_exp2f)
            float ex = __builtin_amdgcn_exp2f((l - Mrow) * LOG2E);
#else
            float ex = exp2f((l - Mrow) * LOG2E);
#endif
            float p = ((aw >> k) & 1u) ? ex : 0.f;
            rsum += p;
            A[e] = (__bf16)p;
        }

        unsigned idx = (unsigned)((((b * JT + jt) * HH + h) * 4) * 32 + lane);
        v16bf B0 = fbase[idx];
        v16bf B1 = fbase[idx + 32];
        v16bf B2 = fbase[idx + 64];
        v16bf B3 = fbase[idx + 96];

        acc0 = __builtin_amdgcn_wmma_f32_16x16x32_bf16(false, A, false, B0, (short)0, acc0, false, false);
        acc1 = __builtin_amdgcn_wmma_f32_16x16x32_bf16(false, A, false, B1, (short)0, acc1, false, false);
        acc2 = __builtin_amdgcn_wmma_f32_16x16x32_bf16(false, A, false, B2, (short)0, acc2, false, false);
        acc3 = __builtin_amdgcn_wmma_f32_16x16x32_bf16(false, A, false, B3, (short)0, acc3, false, false);
    }

    // row sums: halves cover disjoint K slots of the same row
    rsum += __shfl_xor(rsum, 16, 32);
    if (half == 0) s_lds[h * 16 + m] = rsum;
    __syncthreads();

#pragma unroll
    for (int r = 0; r < 8; ++r) {
        int   orow = i0 + r + 8 * half;
        float inv  = 1.0f / s_lds[h * 16 + r + 8 * half];
        float* op  = out + (size_t)(b * NN + orow) * COUT + h * CH + m;
        op[0]  = acc0[r] * inv;
        op[16] = acc1[r] * inv;
        op[32] = acc2[r] * inv;
        op[48] = acc3[r] * inv;
    }
}

// ---------------------------------------------------------------------------
extern "C" void kernel_launch(void* const* d_in, const int* in_sizes, int n_in,
                              void* d_out, int out_size, void* d_ws, size_t ws_size,
                              hipStream_t stream) {
    (void)in_sizes; (void)n_in; (void)out_size; (void)ws_size;
    const float* x    = (const float*)d_in[0];   // node_feats (B,N,256)
    const int*   adj  = (const int*)d_in[1];     // adj_matrix (B,N,N)
    const float* W    = (const float*)d_in[2];   // (256,256)
    const float* bias = (const float*)d_in[3];   // (256,)
    const float* a    = (const float*)d_in[4];   // (H, 2*CH)
    float*       out  = (float*)d_out;

    char* ws = (char*)d_ws;
    size_t off = 0;
    float* feats = (float*)(ws + off);           off += (size_t)BB * NN * COUT * 4;      // 8 MB
    float* lsT   = (float*)(ws + off);           off += (size_t)BB * HH * NN * 4;        // 128 KB
    float* ldT   = (float*)(ws + off);           off += (size_t)BB * HH * NN * 4;        // 128 KB
    float* gmx   = (float*)(ws + off);           off += 256;
    unsigned short* fsw = (unsigned short*)(ws + off); off += (size_t)BB * JT * HH * 4 * 32 * 8 * 2 * 2; // 4 MB
    unsigned* abits = (unsigned*)(ws + off);     off += (size_t)BB * NN * JT * 4;        // 2 MB

    feats_gemm  <<<1024, 256, 0, stream>>>(x, W, bias, feats);
    logits_kernel<<<(BB * NN * HH) / 256, 256, 0, stream>>>(feats, a, lsT, ldT);
    gmax_kernel <<<BB * HH, 256, 0, stream>>>(ldT, gmx);
    swz_kernel  <<<(BB * JT * HH * 4 * 32 * 8) / 256, 256, 0, stream>>>(feats, fsw);
    adjbits_kernel<<<(BB * NN * JT) / 256, 256, 0, stream>>>(adj, abits);
    attn_kernel <<<BB * (NN / 16), 128, 0, stream>>>(abits, lsT, ldT, gmx, fsw, out);
}
